// MultiHeadAttention_85478439125982
// MI455X (gfx1250) — compile-verified, matrix operand order checked by reading, not run on hardware
//
#include <hip/hip_runtime.h>

// ---------------------------------------------------------------------------
// Causal MHA forward for gfx1250 (MI455X), bf16 WMMA pipeline, v3:
// ping-pong double buffering (no rotation copies) in all WMMA loops.
//   B=4, S=2048, E=1024, H=16, D=64
// ---------------------------------------------------------------------------

typedef __attribute__((ext_vector_type(16))) __bf16 v16bf;
typedef __attribute__((ext_vector_type(8)))  __bf16 v8bf;
typedef __attribute__((ext_vector_type(8)))  float  v8f;

constexpr int Bc = 4;
constexpr int Sc = 2048;
constexpr int Ec = 1024;
constexpr int Hc = 16;
constexpr int Dc = 64;

__device__ __forceinline__ v8f wmma_bf16(v16bf a, v16bf b, v8f c) {
  return __builtin_amdgcn_wmma_f32_16x16x32_bf16(
      /*neg_a=*/false, a, /*neg_b=*/false, b,
      /*c_mod=*/(short)0, c, /*reuse_a=*/false, /*reuse_b=*/false);
}

// 16-bit A/B fragment loader (ISA 7.12.2 layout). Source must be N-major
// (row-major for A, B^T row-major for B) -> two contiguous global_load_b128.
__device__ __forceinline__ v16bf load_frag(const __bf16* __restrict__ base, int ld) {
  const int lane = threadIdx.x & 31;
  const int m    = lane & 15;
  const int half = lane >> 4;
  const __bf16* p = base + (size_t)m * ld + half * 8;
  v8bf lo = *reinterpret_cast<const v8bf*>(p);
  v8bf hi = *reinterpret_cast<const v8bf*>(p + 16);
  v16bf r;
#pragma unroll
  for (int i = 0; i < 8; ++i) { r[i] = lo[i]; r[i + 8] = hi[i]; }
  return r;
}

// C-tile store: lane (nl,half), vgpr r -> element (m = half*8+r, n = nl)
__device__ __forceinline__ void store_c_bf16(__bf16* base, int ld, v8f c,
                                             int nl, int half) {
#pragma unroll
  for (int r = 0; r < 8; ++r)
    base[(size_t)(half * 8 + r) * ld + nl] = (__bf16)c[r];
}

// ---------------------------------------------------------------------------
// Prep kernels
__global__ void k_cvt_bf16(const float* __restrict__ src, __bf16* __restrict__ dst, int n) {
  int i = blockIdx.x * blockDim.x + threadIdx.x;
  if (i < n) dst[i] = (__bf16)src[i];
}

__global__ void k_cvt_w_tr(const float* __restrict__ src, __bf16* __restrict__ dst) {
  int i = blockIdx.x * blockDim.x + threadIdx.x;   // over H*D*E
  if (i >= Hc * Dc * Ec) return;
  int e = i % Ec;
  int t = i / Ec;
  int d = t % Dc;
  int h = t / Dc;
  dst[i] = (__bf16)src[((size_t)h * Ec + e) * Dc + d];
}

// ---------------------------------------------------------------------------
// QKV projection. One wave: 32(row) x 64(D) tile of one of {q,k,v}.
// Ping-pong buffered K loop (step 64 = 2 WMMA k-steps), no rotation copies.
__global__ void __launch_bounds__(128) k_qkv(
    const __bf16* __restrict__ xb,
    const __bf16* __restrict__ wqt, const __bf16* __restrict__ wkt,
    const __bf16* __restrict__ wvt,
    __bf16* __restrict__ q, __bf16* __restrict__ k, __bf16* __restrict__ vt) {
  const int wid = blockIdx.x * (blockDim.x >> 5) + (threadIdx.x >> 5);
  constexpr int RT = Bc * Sc / 32;                 // 256 row tiles (32 rows)
  const int rt  = wid % RT;
  const int rem = wid / RT;
  const int h   = rem % Hc;
  const int mat = rem / Hc;                        // 0=q 1=k 2=v (wave-uniform)
  const __bf16* wt = (mat == 0) ? wqt : (mat == 1) ? wkt : wvt;

  const int lane = threadIdx.x & 31;
  const int nl = lane & 15, half = lane >> 4;

  const __bf16* arow = xb + (size_t)rt * 32 * Ec;
  const __bf16* brow = wt + (size_t)h * Dc * Ec;

  v8f acc[2][4] = {};
  v16bf A0[2], B0[4], A1[2], B1[4];
  A0[0] = load_frag(arow, Ec);
  A0[1] = load_frag(arow + (size_t)16 * Ec, Ec);
#pragma unroll
  for (int nt = 0; nt < 4; ++nt)
    B0[nt] = load_frag(brow + (size_t)nt * 16 * Ec, Ec);

  for (int kk = 0; kk < Ec; kk += 64) {
    // prefetch kk+32 into buffer 1 (always in range: kk+32 <= Ec-32)
    A1[0] = load_frag(arow + kk + 32, Ec);
    A1[1] = load_frag(arow + (size_t)16 * Ec + kk + 32, Ec);
#pragma unroll
    for (int nt = 0; nt < 4; ++nt)
      B1[nt] = load_frag(brow + (size_t)nt * 16 * Ec + kk + 32, Ec);
    // consume buffer 0 (holds kk)
#pragma unroll
    for (int nt = 0; nt < 4; ++nt) {
      acc[0][nt] = wmma_bf16(A0[0], B0[nt], acc[0][nt]);
      acc[1][nt] = wmma_bf16(A0[1], B0[nt], acc[1][nt]);
    }
    // prefetch kk+64 into buffer 0 (clamped on last iteration)
    const int kn = (kk + 64 < Ec) ? (kk + 64) : 0;
    A0[0] = load_frag(arow + kn, Ec);
    A0[1] = load_frag(arow + (size_t)16 * Ec + kn, Ec);
#pragma unroll
    for (int nt = 0; nt < 4; ++nt)
      B0[nt] = load_frag(brow + (size_t)nt * 16 * Ec + kn, Ec);
    // consume buffer 1 (holds kk+32)
#pragma unroll
    for (int nt = 0; nt < 4; ++nt) {
      acc[0][nt] = wmma_bf16(A1[0], B1[nt], acc[0][nt]);
      acc[1][nt] = wmma_bf16(A1[1], B1[nt], acc[1][nt]);
    }
  }

  const int b  = (rt * 32) / Sc;
  const int s0 = (rt * 32) % Sc;
  if (mat < 2) {
    __bf16* dst = ((mat == 0) ? q : k) + ((size_t)(b * Hc + h) * Sc + s0) * Dc;
#pragma unroll
    for (int ms = 0; ms < 2; ++ms)
#pragma unroll
      for (int nt = 0; nt < 4; ++nt)
        store_c_bf16(dst + (size_t)ms * 16 * Dc + nt * 16, Dc, acc[ms][nt], nl, half);
  } else {
    __bf16* dst = vt + (size_t)(b * Hc + h) * Dc * Sc;
#pragma unroll
    for (int ms = 0; ms < 2; ++ms)
#pragma unroll
      for (int nt = 0; nt < 4; ++nt)
#pragma unroll
        for (int r = 0; r < 8; ++r)
          dst[(size_t)(nt * 16 + nl) * Sc + s0 + ms * 16 + half * 8 + r] =
              (__bf16)acc[ms][nt][r];
  }
}

// ---------------------------------------------------------------------------
// Flash-style causal attention, one wave per (b,h, 16-row tile).
// Chunk-pair loop with ping-pong K-fragment buffers; chunk count rounded up to
// even -- a padded chunk is fully causally masked and is an exact no-op
// (alpha=1, p=0) while all its loads remain in bounds (nc2*32 <= S).
__global__ void __launch_bounds__(128) k_attn(
    const __bf16* __restrict__ q, const __bf16* __restrict__ kmat,
    const __bf16* __restrict__ vt, __bf16* __restrict__ ao) {
  __shared__ __bf16 lds[4][16][40];                // per-wave 16x32 P tile (+pad)
  const int widx = threadIdx.x >> 5;
  const int wid  = blockIdx.x * 4 + widx;
  const int st = wid % (Sc / 16);
  const int bh = wid / (Sc / 16);                  // b*H + h
  const int s0 = st * 16;
  const int lane = threadIdx.x & 31;
  const int nl = lane & 15, half = lane >> 4;

  const __bf16* qbase = q + ((size_t)bh * Sc + s0) * Dc;
  v16bf a0 = load_frag(qbase, Dc);
  v16bf a1 = load_frag(qbase + 32, Dc);

  float mrow[8], lsum[8];
#pragma unroll
  for (int r = 0; r < 8; ++r) { mrow[r] = -1e30f; lsum[r] = 0.0f; }
  v8f acc[4] = {};
  const float scale = 0.125f;                      // 1/sqrt(64)

  const __bf16* kbh = kmat + (size_t)bh * Sc * Dc;
  const __bf16* vbh = vt + (size_t)bh * Dc * Sc;

  // one 32-column chunk: scores (consume K buffer), V loads, prefetch next K
  // chunk into the other buffer, online softmax, LDS relayout, PV accumulate.
  auto chunk = [&](int t0, v16bf& K0, v16bf& K1, v16bf& K2, v16bf& K3,
                   v16bf& N0, v16bf& N1, v16bf& N2, v16bf& N3, int tn) {
    v8f sc0 = {}, sc1 = {};
    sc0 = wmma_bf16(a0, K0, sc0);
    sc0 = wmma_bf16(a1, K1, sc0);
    sc1 = wmma_bf16(a0, K2, sc1);
    sc1 = wmma_bf16(a1, K3, sc1);
    // V loads for this chunk (consumed after softmax math)
    v16bf VF[4];
#pragma unroll
    for (int nt = 0; nt < 4; ++nt)
      VF[nt] = load_frag(vbh + (size_t)nt * 16 * Sc + t0, Sc);
    // prefetch next chunk's K fragments into the other buffer
    const __bf16* kb = kbh + (size_t)tn * Dc;
    N0 = load_frag(kb, Dc);
    N1 = load_frag(kb + 32, Dc);
    N2 = load_frag(kb + (size_t)16 * Dc, Dc);
    N3 = load_frag(kb + (size_t)16 * Dc + 32, Dc);
    // causal mask + scale + chunk max
    float p[2][8], cmax[8];
#pragma unroll
    for (int r = 0; r < 8; ++r) {
      const int row = s0 + half * 8 + r;
      float v0 = (t0 + nl      <= row) ? sc0[r] * scale : -1e30f;
      float v1 = (t0 + 16 + nl <= row) ? sc1[r] * scale : -1e30f;
      p[0][r] = v0; p[1][r] = v1;
      cmax[r] = fmaxf(v0, v1);
    }
#pragma unroll
    for (int r = 0; r < 8; ++r)
#pragma unroll
      for (int m = 1; m < 16; m <<= 1)
        cmax[r] = fmaxf(cmax[r], __shfl_xor(cmax[r], m, 32));
    float alpha[8], rs[8];
#pragma unroll
    for (int r = 0; r < 8; ++r) {
      float nm = fmaxf(mrow[r], cmax[r]);
      alpha[r] = __expf(mrow[r] - nm);
      mrow[r] = nm;
      float e0 = __expf(p[0][r] - nm);
      float e1 = __expf(p[1][r] - nm);
      p[0][r] = e0; p[1][r] = e1;
      rs[r] = e0 + e1;
    }
#pragma unroll
    for (int r = 0; r < 8; ++r) {
#pragma unroll
      for (int m = 1; m < 16; m <<= 1)
        rs[r] += __shfl_xor(rs[r], m, 32);
      lsum[r] = lsum[r] * alpha[r] + rs[r];
    }
#pragma unroll
    for (int nt = 0; nt < 4; ++nt)
#pragma unroll
      for (int r = 0; r < 8; ++r)
        acc[nt][r] *= alpha[r];
    // P: C layout -> LDS -> A layout
#pragma unroll
    for (int u = 0; u < 2; ++u)
#pragma unroll
      for (int r = 0; r < 8; ++r)
        lds[widx][half * 8 + r][u * 16 + nl] = (__bf16)p[u][r];
    asm volatile("s_wait_dscnt 0" ::: "memory");
    v16bf pa;
    {
      const __bf16* lp = &lds[widx][nl][half * 8]; // 16B aligned (stride 80B)
      v8bf lo = *reinterpret_cast<const v8bf*>(lp);
      v8bf hi = *reinterpret_cast<const v8bf*>(lp + 16);
#pragma unroll
      for (int i = 0; i < 8; ++i) { pa[i] = lo[i]; pa[i + 8] = hi[i]; }
    }
#pragma unroll
    for (int nt = 0; nt < 4; ++nt)
      acc[nt] = wmma_bf16(pa, VF[nt], acc[nt]);
  };

  const int nchunk = st / 2 + 1;                   // exact 32-col chunk count
  const int nc2    = (nchunk + 1) & ~1;            // rounded up to even

  // prime chunk 0 into buffer A
  v16bf KA0 = load_frag(kbh, Dc);
  v16bf KA1 = load_frag(kbh + 32, Dc);
  v16bf KA2 = load_frag(kbh + (size_t)16 * Dc, Dc);
  v16bf KA3 = load_frag(kbh + (size_t)16 * Dc + 32, Dc);
  v16bf KB0, KB1, KB2, KB3;

  for (int c = 0; c < nc2; c += 2) {               // wave-uniform
    const int tn2 = (c + 2 < nc2) ? (c + 2) * 32 : 0;
    chunk((c + 0) * 32, KA0, KA1, KA2, KA3, KB0, KB1, KB2, KB3, (c + 1) * 32);
    chunk((c + 1) * 32, KB0, KB1, KB2, KB3, KA0, KA1, KA2, KA3, tn2);
  }

  // normalize + store attn-out [B,S,H*D] bf16
  const int b = bh / Hc, h = bh % Hc;
  __bf16* dst = ao + (size_t)(b * Sc + s0) * (Hc * Dc) + h * Dc;
#pragma unroll
  for (int nt = 0; nt < 4; ++nt)
#pragma unroll
    for (int r = 0; r < 8; ++r)
      dst[(size_t)(half * 8 + r) * (Hc * Dc) + nt * 16 + nl] =
          (__bf16)(acc[nt][r] * (1.0f / lsum[r]));
}

// ---------------------------------------------------------------------------
// Output projection: 32x32 tile per wave, ping-pong buffered K loop.
__global__ void __launch_bounds__(128) k_outproj(
    const __bf16* __restrict__ ao, const __bf16* __restrict__ wob,
    const float* __restrict__ bo, float* __restrict__ out) {
  const int wid = blockIdx.x * (blockDim.x >> 5) + (threadIdx.x >> 5);
  const int ct = wid % (Ec / 32);
  const int rt = wid / (Ec / 32);
  const int lane = threadIdx.x & 31;
  const int nl = lane & 15, half = lane >> 4;

  const __bf16* arow = ao  + (size_t)rt * 32 * Ec;
  const __bf16* brow = wob + (size_t)ct * 32 * Ec;

  v8f acc[2][2] = {};
  v16bf A0[2], B0[2], A1[2], B1[2];
  A0[0] = load_frag(arow, Ec);
  A0[1] = load_frag(arow + (size_t)16 * Ec, Ec);
  B0[0] = load_frag(brow, Ec);
  B0[1] = load_frag(brow + (size_t)16 * Ec, Ec);

  for (int kk = 0; kk < Ec; kk += 64) {
    A1[0] = load_frag(arow + kk + 32, Ec);
    A1[1] = load_frag(arow + (size_t)16 * Ec + kk + 32, Ec);
    B1[0] = load_frag(brow + kk + 32, Ec);
    B1[1] = load_frag(brow + (size_t)16 * Ec + kk + 32, Ec);
#pragma unroll
    for (int ms = 0; ms < 2; ++ms)
#pragma unroll
      for (int ns = 0; ns < 2; ++ns)
        acc[ms][ns] = wmma_bf16(A0[ms], B0[ns], acc[ms][ns]);
    const int kn = (kk + 64 < Ec) ? (kk + 64) : 0;
    A0[0] = load_frag(arow + kn, Ec);
    A0[1] = load_frag(arow + (size_t)16 * Ec + kn, Ec);
    B0[0] = load_frag(brow + kn, Ec);
    B0[1] = load_frag(brow + (size_t)16 * Ec + kn, Ec);
#pragma unroll
    for (int ms = 0; ms < 2; ++ms)
#pragma unroll
      for (int ns = 0; ns < 2; ++ns)
        acc[ms][ns] = wmma_bf16(A1[ms], B1[ns], acc[ms][ns]);
  }

#pragma unroll
  for (int ns = 0; ns < 2; ++ns) {
    const float bias = bo[ct * 32 + ns * 16 + nl];
#pragma unroll
    for (int ms = 0; ms < 2; ++ms) {
      float* dst = out + (size_t)(rt * 32 + ms * 16) * Ec + ct * 32 + ns * 16;
#pragma unroll
      for (int r = 0; r < 8; ++r)
        dst[(size_t)(half * 8 + r) * Ec + nl] = acc[ms][ns][r] + bias;
    }
  }
}

// ---------------------------------------------------------------------------
extern "C" void kernel_launch(void* const* d_in, const int* in_sizes, int n_in,
                              void* d_out, int out_size, void* d_ws, size_t ws_size,
                              hipStream_t stream) {
  (void)in_sizes; (void)n_in; (void)out_size; (void)ws_size;
  const float* x  = (const float*)d_in[0];
  const float* Wq = (const float*)d_in[1];
  const float* Wk = (const float*)d_in[2];
  const float* Wv = (const float*)d_in[3];
  const float* Wo = (const float*)d_in[4];
  const float* bo = (const float*)d_in[5];
  float* out = (float*)d_out;

  // Workspace layout (bytes)
  char* ws = (char*)d_ws;
  const size_t xbB = (size_t)Bc * Sc * Ec * 2;       // 16 MB (reused for ao)
  const size_t wtB = (size_t)Hc * Dc * Ec * 2;       //  2 MB each
  const size_t woB = (size_t)Ec * Ec * 2;            //  2 MB
  const size_t qB  = (size_t)Bc * Hc * Sc * Dc * 2;  // 16 MB each
  __bf16* xb  = (__bf16*)(ws);
  __bf16* ao  = xb;                                  // alias: xb dead after k_qkv
  __bf16* wqt = (__bf16*)(ws + xbB);
  __bf16* wkt = (__bf16*)(ws + xbB + wtB);
  __bf16* wvt = (__bf16*)(ws + xbB + 2 * wtB);
  __bf16* wob = (__bf16*)(ws + xbB + 3 * wtB);
  __bf16* qb  = (__bf16*)(ws + xbB + 3 * wtB + woB);
  __bf16* kb  = (__bf16*)(ws + xbB + 3 * wtB + woB + qB);
  __bf16* vtb = (__bf16*)(ws + xbB + 3 * wtB + woB + 2 * qB);

  {
    int n = Bc * Sc * Ec;
    k_cvt_bf16<<<(n + 255) / 256, 256, 0, stream>>>(x, xb, n);
    int nw = Hc * Dc * Ec;
    k_cvt_w_tr<<<(nw + 255) / 256, 256, 0, stream>>>(Wq, wqt);
    k_cvt_w_tr<<<(nw + 255) / 256, 256, 0, stream>>>(Wk, wkt);
    k_cvt_w_tr<<<(nw + 255) / 256, 256, 0, stream>>>(Wv, wvt);
    int no = Ec * Ec;
    k_cvt_bf16<<<(no + 255) / 256, 256, 0, stream>>>(Wo, wob, no);
  }
  {  // QKV: 256 row-tiles * 16 heads * 3 matrices waves, 4 waves/block
    int waves = (Bc * Sc / 32) * Hc * 3;
    k_qkv<<<waves / 4, 128, 0, stream>>>(xb, wqt, wkt, wvt, qb, kb, vtb);
  }
  {  // attention: B*H*(S/16) = 8192 waves
    int waves = Bc * Hc * (Sc / 16);
    k_attn<<<waves / 4, 128, 0, stream>>>(qb, kb, vtb, ao);
  }
  {  // output projection: 256 * 32 waves
    int waves = (Bc * Sc / 32) * (Ec / 32);
    k_outproj<<<waves / 4, 128, 0, stream>>>(ao, wob, bo, out);
  }
}
